// GRUCell_42236708388937
// MI455X (gfx1250) — compile-verified, matrix operand order checked by reading, not run on hardware
//
#include <hip/hip_runtime.h>
#include <hip/hip_bf16.h>

// Problem sizes (fixed by the reference)
#define BATCH 65536
#define INDIM 512
#define HDIM  512
#define KCAT  1024   // concat K = IN + H
#define NALL  1536   // 3*H gate rows

typedef __attribute__((ext_vector_type(16))) _Float16 v16h;
typedef __attribute__((ext_vector_type(8)))  _Float16 v8h;
typedef __attribute__((ext_vector_type(4)))  _Float16 v4h;
typedef __attribute__((ext_vector_type(8)))  float    v8f;
typedef __attribute__((ext_vector_type(4)))  int      v4i;

#define LDST 40   // LDS row stride in f16 (32 data + 8 pad) -> 80B, bank-conflict-free b128 frags

// ---- CDNA5 async global->LDS path (ASYNCcnt), with guaranteed fallback -----
#if defined(__gfx1250__) && __has_builtin(__builtin_amdgcn_global_load_async_to_lds_b128)
#define ASYNC_LDS 1
#else
#define ASYNC_LDS 0
#endif

#if ASYNC_LDS
// signature (from clang diagnostic): (AS1 int4*, AS3 int4*, imm offset, imm cpol)
typedef __attribute__((address_space(1))) v4i gv4i;
typedef __attribute__((address_space(3))) v4i lv4i;
#define GLD_ASYNC(gp, lp)                                                     \
    __builtin_amdgcn_global_load_async_to_lds_b128((gv4i*)(gp), (lv4i*)(lp), 0, 0)
#if __has_builtin(__builtin_amdgcn_s_wait_asynccnt)
#define ASYNC_WAIT() __builtin_amdgcn_s_wait_asynccnt(0)
#else
#define ASYNC_WAIT() asm volatile("s_wait_asynccnt 0x0" ::: "memory")
#endif
#endif

// ---------------------------------------------------------------------------
// Kernel 0: convert weights to f16 concat layout, fuse biases.
// Wc[n][k] : k<512 -> W_x2h[n][k], else W_h2h[n][k-512].  (1536 x 1024)
// ---------------------------------------------------------------------------
__global__ void prep_weights(const float* __restrict__ Wx, const float* __restrict__ Wh,
                             const float* __restrict__ bx, const float* __restrict__ bh,
                             _Float16* __restrict__ Wc, float* __restrict__ bias_ru,
                             float* __restrict__ bias_xc, float* __restrict__ bias_hc) {
    int t = blockIdx.x * blockDim.x + threadIdx.x;   // 1536*1024 threads
    int n = t >> 10;
    int k = t & 1023;
    float w = (k < 512) ? Wx[n * 512 + k] : Wh[n * 512 + (k - 512)];
    Wc[t] = (_Float16)w;
    if (t < 1024) {
        bias_ru[t] = bx[t] + bh[t];
    } else if (t < 1536) {
        bias_xc[t - 1024] = bx[t];
        bias_hc[t - 1024] = bh[t];
    }
}

// ---------------------------------------------------------------------------
// Kernel 1: attention + h_pre, one wave (32 lanes) per batch row.
// att0 = sigmoid((a-b)·wh)   (x·wx and b_mlp cancel in the 2-way softmax)
// Writes: att pair to d_out tail, [x|h_pre] as f16 row (B x 1024) for GEMMs.
// ---------------------------------------------------------------------------
__global__ __launch_bounds__(256) void attn_prep(
        const float* __restrict__ x, const float* __restrict__ a,
        const float* __restrict__ b, const float* __restrict__ wmlp,
        _Float16* __restrict__ xh, float* __restrict__ att_out) {
    const int lane = threadIdx.x & 31;
    const int wave = threadIdx.x >> 5;
    const size_t row = (size_t)blockIdx.x * 8 + wave;

    const float4* ap = (const float4*)(a + row * HDIM);
    const float4* bp = (const float4*)(b + row * HDIM);
    const float4* xp = (const float4*)(x + row * INDIM);
    const float4* wp = (const float4*)(wmlp + INDIM);   // wh part of W_mlp row

    float4 av[4], bv[4], xv[4];
    float d = 0.f;
#pragma unroll
    for (int i = 0; i < 4; ++i) {
        int c4 = i * 32 + lane;                 // float4 index; col = 4*c4
        av[i] = ap[c4]; bv[i] = bp[c4]; xv[i] = xp[c4];
        float4 wv = wp[c4];
        d += (av[i].x - bv[i].x) * wv.x + (av[i].y - bv[i].y) * wv.y +
             (av[i].z - bv[i].z) * wv.z + (av[i].w - bv[i].w) * wv.w;
    }
#pragma unroll
    for (int off = 16; off > 0; off >>= 1) d += __shfl_xor(d, off, 32);

    float att0 = 1.f / (1.f + __expf(-d));
    float att1 = 1.f - att0;

    _Float16* xrow = xh + row * KCAT;
#pragma unroll
    for (int i = 0; i < 4; ++i) {
        int c4 = i * 32 + lane;
        v4h hx = {(_Float16)xv[i].x, (_Float16)xv[i].y, (_Float16)xv[i].z, (_Float16)xv[i].w};
        *(v4h*)(xrow + c4 * 4) = hx;
        v4h hv = {(_Float16)(att0 * av[i].x + att1 * bv[i].x),
                  (_Float16)(att0 * av[i].y + att1 * bv[i].y),
                  (_Float16)(att0 * av[i].z + att1 * bv[i].z),
                  (_Float16)(att0 * av[i].w + att1 * bv[i].w)};
        *(v4h*)(xrow + INDIM + c4 * 4) = hv;
    }
    if (lane == 0) {
        att_out[row * 2 + 0] = att0;
        att_out[row * 2 + 1] = att1;
    }
}

// Build a v16h WMMA fragment from two 16-byte LDS chunks.
// 16-bit operand layout: lanes 0-15 hold row (lane&15), K {0..7,16..23};
// lanes 16-31 hold K {8..15,24..31}  ->  two b128 reads at +lh*16B and +32B.
__device__ __forceinline__ v16h frag_from_lds(const _Float16* base, int r, int lh) {
    v8h lo = *(const v8h*)(base + r * LDST + lh * 8);
    v8h hi = *(const v8h*)(base + r * LDST + 16 + lh * 8);
    return __builtin_shufflevector(lo, hi, 0, 1, 2, 3, 4, 5, 6, 7, 8, 9, 10, 11, 12, 13, 14, 15);
}

// ---------------------------------------------------------------------------
// Kernel 2: GEMM for r,u gates.  C[M=65536, N=1024] = A[M,1024] * Wc[0:1024,:]^T
// BM=128, BN=128, KC-step 32, double-buffered LDS, 8 waves (4 along M x 2 along N).
// Tile feed: async global->LDS (ASYNCcnt) when available.
// Epilogue: sigmoid(acc + bias) -> r (n<512) / u (n>=512) as f16.
// ---------------------------------------------------------------------------
__global__ __launch_bounds__(256) void gemm_ru(
        const _Float16* __restrict__ A, const _Float16* __restrict__ W,
        const float* __restrict__ bias,
        _Float16* __restrict__ rbuf, _Float16* __restrict__ ubuf) {
    __shared__ _Float16 ldsA[2][128 * LDST];
    __shared__ _Float16 ldsW[2][128 * LDST];

    const int tid  = threadIdx.x;
    const int lane = tid & 31;
    const int lr   = lane & 15, lh = lane >> 4;
    const int wave = tid >> 5;
    const int waveM = wave & 3, waveN = wave >> 2;
    const size_t mBase = (size_t)blockIdx.y * 128;
    const int    nBase = blockIdx.x * 128;

    // per-thread staging: 2 chunks of A tile, 2 chunks of W tile (16B each)
    const int crow = tid >> 2;       // 0..63
    const int ck   = tid & 3;        // 8-f16 chunk within the 32-wide K slab
    const _Float16* Ag0 = A + (mBase + crow) * KCAT + ck * 8;
    const _Float16* Ag1 = A + (mBase + crow + 64) * KCAT + ck * 8;
    const _Float16* Wg0 = W + (size_t)(nBase + crow) * KCAT + ck * 8;
    const _Float16* Wg1 = W + (size_t)(nBase + crow + 64) * KCAT + ck * 8;
    const int s0 = crow * LDST + ck * 8;
    const int s1 = (crow + 64) * LDST + ck * 8;

    v8f acc[2][4];
#pragma unroll
    for (int mi = 0; mi < 2; ++mi)
#pragma unroll
        for (int ni = 0; ni < 4; ++ni) acc[mi][ni] = (v8f){};

#if ASYNC_LDS
#define ISSUE_RU(nb, kt)                                                      \
    do {                                                                      \
        int ko = (kt) * 32;                                                   \
        GLD_ASYNC(Ag0 + ko, &ldsA[nb][s0]);                                   \
        GLD_ASYNC(Ag1 + ko, &ldsA[nb][s1]);                                   \
        GLD_ASYNC(Wg0 + ko, &ldsW[nb][s0]);                                   \
        GLD_ASYNC(Wg1 + ko, &ldsW[nb][s1]);                                   \
    } while (0)
    ISSUE_RU(0, 0);
    ASYNC_WAIT();
    __syncthreads();
#else
    v8h ra0 = *(const v8h*)Ag0, ra1 = *(const v8h*)Ag1;
    v8h rb0 = *(const v8h*)Wg0, rb1 = *(const v8h*)Wg1;
    *(v8h*)&ldsA[0][s0] = ra0; *(v8h*)&ldsA[0][s1] = ra1;
    *(v8h*)&ldsW[0][s0] = rb0; *(v8h*)&ldsW[0][s1] = rb1;
    __syncthreads();
#endif

    int buf = 0;
    for (int kt = 0; kt < 32; ++kt) {
#if ASYNC_LDS
        // buf^1 was last read in iteration kt-1; the barrier that ended kt-1
        // retired those reads, so its LDS space is free for the next tile.
        if (kt + 1 < 32) ISSUE_RU(buf ^ 1, kt + 1);
#else
        if (kt + 1 < 32) {                 // prefetch next K-slab into regs
            int ko = (kt + 1) * 32;
            ra0 = *(const v8h*)(Ag0 + ko); ra1 = *(const v8h*)(Ag1 + ko);
            rb0 = *(const v8h*)(Wg0 + ko); rb1 = *(const v8h*)(Wg1 + ko);
        }
#endif
        const _Float16* As = &ldsA[buf][0];
        const _Float16* Bs = &ldsW[buf][0];
        v16h af[2], bf[4];
#pragma unroll
        for (int mi = 0; mi < 2; ++mi) af[mi] = frag_from_lds(As, waveM * 32 + mi * 16 + lr, lh);
#pragma unroll
        for (int ni = 0; ni < 4; ++ni) bf[ni] = frag_from_lds(Bs, waveN * 64 + ni * 16 + lr, lh);
#pragma unroll
        for (int mi = 0; mi < 2; ++mi)
#pragma unroll
            for (int ni = 0; ni < 4; ++ni)
                acc[mi][ni] = __builtin_amdgcn_wmma_f32_16x16x32_f16(
                    false, af[mi], false, bf[ni], (short)0, acc[mi][ni], false, false);
        if (kt + 1 < 32) {
#if ASYNC_LDS
            ASYNC_WAIT();
            __syncthreads();
#else
            __syncthreads();
            int nb = buf ^ 1;
            *(v8h*)&ldsA[nb][s0] = ra0; *(v8h*)&ldsA[nb][s1] = ra1;
            *(v8h*)&ldsW[nb][s0] = rb0; *(v8h*)&ldsW[nb][s1] = rb1;
            __syncthreads();
#endif
        }
        buf ^= 1;
    }

    // epilogue: C layout -> VGPR j: lanes 0-15 M=j, lanes 16-31 M=8+j; N=lane&15
#pragma unroll
    for (int mi = 0; mi < 2; ++mi) {
        size_t rowb = mBase + waveM * 32 + mi * 16 + lh * 8;
#pragma unroll
        for (int ni = 0; ni < 4; ++ni) {
            int n = nBase + waveN * 64 + ni * 16 + lr;
            float bs = bias[n];
            _Float16* dst = (n < 512) ? (rbuf + n) : (ubuf + (n - 512));
#pragma unroll
            for (int j = 0; j < 8; ++j) {
                float v = acc[mi][ni][j] + bs;
                dst[(rowb + j) * 512] = (_Float16)(1.f / (1.f + __expf(-v)));
            }
        }
    }
}

// ---------------------------------------------------------------------------
// Kernel 3: candidate GEMM + full GRU epilogue.
// Two accumulator sets: K-half 0 (x part, W_x2h c-rows) and K-half 1 (h part).
// BM=128, BN=64; 8 waves along M.  c = tanh(xc + r*hc); h_t = c + u*(h_pre - c).
// ---------------------------------------------------------------------------
__global__ __launch_bounds__(256) void gemm_c(
        const _Float16* __restrict__ A, const _Float16* __restrict__ W,  // W = Wc + 1024*KCAT
        const float* __restrict__ bias_xc, const float* __restrict__ bias_hc,
        const _Float16* __restrict__ rbuf, const _Float16* __restrict__ ubuf,
        float* __restrict__ out) {
    __shared__ _Float16 ldsA[2][128 * LDST];
    __shared__ _Float16 ldsW[2][64 * LDST];

    const int tid  = threadIdx.x;
    const int lane = tid & 31;
    const int lr   = lane & 15, lh = lane >> 4;
    const int wave = tid >> 5;                 // waveM = wave (16 rows each)
    const size_t mBase = (size_t)blockIdx.y * 128;
    const int    nBase = blockIdx.x * 64;

    const int crow = tid >> 2;
    const int ck   = tid & 3;
    const _Float16* Ag0 = A + (mBase + crow) * KCAT + ck * 8;
    const _Float16* Ag1 = A + (mBase + crow + 64) * KCAT + ck * 8;
    const _Float16* Wg0 = W + (size_t)(nBase + crow) * KCAT + ck * 8;   // crow<64 only
    const int s0 = crow * LDST + ck * 8;
    const int s1 = (crow + 64) * LDST + ck * 8;
    const bool doW = (crow < 64);

    v8f accX[4], accH[4];
#pragma unroll
    for (int ni = 0; ni < 4; ++ni) { accX[ni] = (v8f){}; accH[ni] = (v8f){}; }

#if ASYNC_LDS
#define ISSUE_C(nb, kt)                                                       \
    do {                                                                      \
        int ko = (kt) * 32;                                                   \
        GLD_ASYNC(Ag0 + ko, &ldsA[nb][s0]);                                   \
        GLD_ASYNC(Ag1 + ko, &ldsA[nb][s1]);                                   \
        if (doW) GLD_ASYNC(Wg0 + ko, &ldsW[nb][s0]);                          \
    } while (0)
    ISSUE_C(0, 0);
    ASYNC_WAIT();
    __syncthreads();
#else
    v8h ra0 = *(const v8h*)Ag0, ra1 = *(const v8h*)Ag1;
    v8h rb0 = doW ? *(const v8h*)Wg0 : (v8h){};
    *(v8h*)&ldsA[0][s0] = ra0; *(v8h*)&ldsA[0][s1] = ra1;
    if (doW) *(v8h*)&ldsW[0][s0] = rb0;
    __syncthreads();
#endif

    int buf = 0;
    for (int kt = 0; kt < 32; ++kt) {
#if ASYNC_LDS
        if (kt + 1 < 32) ISSUE_C(buf ^ 1, kt + 1);
#else
        if (kt + 1 < 32) {
            int ko = (kt + 1) * 32;
            ra0 = *(const v8h*)(Ag0 + ko); ra1 = *(const v8h*)(Ag1 + ko);
            if (doW) rb0 = *(const v8h*)(Wg0 + ko);
        }
#endif
        const _Float16* As = &ldsA[buf][0];
        const _Float16* Bs = &ldsW[buf][0];
        v16h af = frag_from_lds(As, wave * 16 + lr, lh);
        v16h bf[4];
#pragma unroll
        for (int ni = 0; ni < 4; ++ni) bf[ni] = frag_from_lds(Bs, ni * 16 + lr, lh);
        if (kt < 16) {
#pragma unroll
            for (int ni = 0; ni < 4; ++ni)
                accX[ni] = __builtin_amdgcn_wmma_f32_16x16x32_f16(
                    false, af, false, bf[ni], (short)0, accX[ni], false, false);
        } else {
#pragma unroll
            for (int ni = 0; ni < 4; ++ni)
                accH[ni] = __builtin_amdgcn_wmma_f32_16x16x32_f16(
                    false, af, false, bf[ni], (short)0, accH[ni], false, false);
        }
        if (kt + 1 < 32) {
#if ASYNC_LDS
            ASYNC_WAIT();
            __syncthreads();
#else
            __syncthreads();
            int nb = buf ^ 1;
            *(v8h*)&ldsA[nb][s0] = ra0; *(v8h*)&ldsA[nb][s1] = ra1;
            if (doW) *(v8h*)&ldsW[nb][s0] = rb0;
            __syncthreads();
#endif
        }
        buf ^= 1;
    }

    size_t rowb = mBase + wave * 16 + lh * 8;
#pragma unroll
    for (int ni = 0; ni < 4; ++ni) {
        int n = nBase + ni * 16 + lr;      // 0..511
        float bxc = bias_xc[n];
        float bhc = bias_hc[n];
#pragma unroll
        for (int j = 0; j < 8; ++j) {
            size_t row = rowb + j;
            float r  = (float)rbuf[row * 512 + n];
            float u  = (float)ubuf[row * 512 + n];
            float hp = (float)A[row * KCAT + INDIM + n];   // h_pre (f16) from concat row
            float c  = tanhf(accX[ni][j] + bxc + r * (accH[ni][j] + bhc));
            out[row * 512 + n] = c + u * (hp - c);
        }
    }
}

// ---------------------------------------------------------------------------
// Launch
// ---------------------------------------------------------------------------
extern "C" void kernel_launch(void* const* d_in, const int* in_sizes, int n_in,
                              void* d_out, int out_size, void* d_ws, size_t ws_size,
                              hipStream_t stream) {
    const float* x     = (const float*)d_in[0];
    const float* a_pre = (const float*)d_in[1];
    const float* b_pre = (const float*)d_in[2];
    const float* W_x2h = (const float*)d_in[3];
    const float* b_x2h = (const float*)d_in[4];
    const float* W_h2h = (const float*)d_in[5];
    const float* b_h2h = (const float*)d_in[6];
    const float* W_mlp = (const float*)d_in[7];
    // d_in[8] = b_mlp: cancels in the 2-way softmax, unused.
    float* out = (float*)d_out;

    // workspace carve-up (bytes)
    char* ws = (char*)d_ws;
    const size_t XH_B   = (size_t)BATCH * KCAT * 2;        // 128 MB f16 [x|h_pre]
    const size_t WC_B   = (size_t)NALL * KCAT * 2;         // 3 MB f16 weights
    const size_t BRU_B  = 1024 * 4;
    const size_t BXC_B  = 512 * 4;
    const size_t BHC_B  = 512 * 4;
    const size_t RB_B   = (size_t)BATCH * HDIM * 2;        // 64 MB f16 r
    _Float16* xh      = (_Float16*)(ws);
    _Float16* Wc      = (_Float16*)(ws + XH_B);
    float*    bias_ru = (float*)(ws + XH_B + WC_B);
    float*    bias_xc = (float*)(ws + XH_B + WC_B + BRU_B);
    float*    bias_hc = (float*)(ws + XH_B + WC_B + BRU_B + BXC_B);
    _Float16* rbuf    = (_Float16*)(ws + XH_B + WC_B + BRU_B + BXC_B + BHC_B);
    _Float16* ubuf    = (_Float16*)(ws + XH_B + WC_B + BRU_B + BXC_B + BHC_B + RB_B);

    prep_weights<<<(NALL * KCAT) / 256, 256, 0, stream>>>(
        W_x2h, W_h2h, b_x2h, b_h2h, Wc, bias_ru, bias_xc, bias_hc);

    attn_prep<<<BATCH / 8, 256, 0, stream>>>(
        x, a_pre, b_pre, W_mlp, xh, out + (size_t)BATCH * HDIM);

    gemm_ru<<<dim3(8, BATCH / 128), 256, 0, stream>>>(xh, Wc, bias_ru, rbuf, ubuf);

    gemm_c<<<dim3(8, BATCH / 128), 256, 0, stream>>>(
        xh, Wc + (size_t)1024 * KCAT, bias_xc, bias_hc, rbuf, ubuf, out);
}
